// MACELayer_53807350284829
// MI455X (gfx1250) — matrix operand order repaired and checked by reading, not compiled, not run on hardware
//
#include <hip/hip_runtime.h>

// ---------------------------------------------------------------------------
// MACE layer for gfx1250 (MI455X), wave32, bf16 WMMA 16x16x32, f32 accum.
// ---------------------------------------------------------------------------

#define DEV __device__ __forceinline__

constexpr int kN = 20000;   // nodes
constexpr int kE = 320000;  // edges
constexpr int kF = 64;      // features

typedef __attribute__((ext_vector_type(16))) __bf16 v16bf;
typedef __attribute__((ext_vector_type(8)))  __bf16 v8bf;
typedef __attribute__((ext_vector_type(8)))  float  v8f;

// float -> bf16 (round to nearest even, ignoring NaN edge cases)
DEV unsigned short f2bf(float x) {
  unsigned int u = __float_as_uint(x);
  u += 0x7FFFu + ((u >> 16) & 1u);
  return (unsigned short)(u >> 16);
}

// Build a 16x32 bf16 A-fragment for one 32-wide K chunk (kc) from an LDS row
// of 64 bf16 (row-major tile, row = lane%16). Contiguous K layout (matches B pack):
//   elem i: K = kc*32 + 16*half + i
// i.e. two adjacent 8-element (16B) runs.
DEV v16bf loadA(const unsigned short* lds_row, int kc, int lh) {
  const v8bf lo = *(const v8bf*)(lds_row + kc * 32 + 16 * lh);
  const v8bf hi = *(const v8bf*)(lds_row + kc * 32 + 16 * lh + 8);
  return __builtin_shufflevector(lo, hi, 0, 1, 2, 3, 4, 5, 6, 7,
                                 8, 9, 10, 11, 12, 13, 14, 15);
}

// B fragments are pre-packed so each lane loads 16 contiguous bf16 (32 bytes).
DEV v16bf loadB(const unsigned short* pb, int kc, int nt, int ntiles, int lane) {
  int fidx = kc * ntiles + nt;
  return *(const v16bf*)(pb + ((size_t)(fidx * 32 + lane)) * 16);
}

DEV v8f wmma_bf16(v16bf a, v16bf b, v8f c) {
  return __builtin_amdgcn_wmma_f32_16x16x32_bf16(
      /*neg_a=*/false, a, /*neg_b=*/false, b,
      /*c_mod=*/(short)0, c, /*reuse_a=*/false, /*reuse_b=*/false);
}

// ---------------------------------------------------------------------------
// K0: pack a (K=64 rows, NC cols) row-major f32 weight into wave32 B-fragment
// layout, converting to bf16 and folding in a scale. Fragment element mapping:
//   PB[(fidx*32 + L)*16 + i] = W[(kc*32 + 16*(L/16) + i) * NC + nt*16 + L%16]
// ---------------------------------------------------------------------------
__global__ void pack_b(const float* __restrict__ W, unsigned short* __restrict__ PB,
                       int NC, float scale) {
  int t = blockIdx.x * blockDim.x + threadIdx.x;
  int ntiles = NC / 16;
  int total = 2 * ntiles * 512;
  if (t >= total) return;
  int fidx = t >> 9;
  int within = t & 511;
  int L = within >> 4;
  int i = within & 15;
  int kc = fidx / ntiles, nt = fidx % ntiles;
  int k = kc * 32 + 16 * (L >> 4) + i;
  int n = nt * 16 + (L & 15);
  PB[t] = f2bf(W[(size_t)k * NC + n] * scale);
}

// ---------------------------------------------------------------------------
// K1: h_s = s @ W_in_s * inv_sf ; h_v[c] = v_c @ W_in_v * inv_sf
// One wave per 16-node tile; 4 planes (s, vx, vy, vz); inv_sf folded in pack.
// ---------------------------------------------------------------------------
__global__ __launch_bounds__(32) void node_in_kernel(
    const float* __restrict__ node_feats,
    const unsigned short* __restrict__ pWs,
    const unsigned short* __restrict__ pWv,
    float* __restrict__ h_s, float* __restrict__ h_v) {
  __shared__ unsigned short A[4 * 16 * 64];  // 8KB bf16
  int lane = threadIdx.x;
  int n0 = blockIdx.x * 16;

  for (int t = lane; t < 4096; t += 32) {
    int p = t >> 10;
    int m = (t >> 6) & 15;
    int f = t & 63;
    A[t] = f2bf(node_feats[((size_t)(n0 + m) * 64 + f) * 4 + p]);
  }
  __syncthreads();

  int lh = lane >> 4, mc = lane & 15;
  for (int p = 0; p < 4; ++p) {
    const unsigned short* pw = (p == 0) ? pWs : pWv;
    float* outp = (p == 0) ? h_s : (h_v + (size_t)(p - 1) * kN * 64);
    v16bf a0 = loadA(&A[p * 1024 + mc * 64], 0, lh);
    v16bf a1 = loadA(&A[p * 1024 + mc * 64], 1, lh);
    for (int nt = 0; nt < 4; ++nt) {
      v8f c = {};
      c = wmma_bf16(a0, loadB(pw, 0, nt, 4, lane), c);
      c = wmma_bf16(a1, loadB(pw, 1, nt, 4, lane), c);
      for (int r = 0; r < 8; ++r) {
        int M = r + 8 * lh;
        outp[(size_t)(n0 + M) * 64 + nt * 16 + mc] = c[r];
      }
    }
  }
}

// ---------------------------------------------------------------------------
// K2: per-species residuals (scalar; per-row weight selection breaks tiling).
// res = einsum('nf,ngf->ng') * inv_sf ; plane p in {s, vx, vy, vz}.
// ---------------------------------------------------------------------------
__global__ void residual_kernel(const float* __restrict__ node_feats,
                                const int* __restrict__ specie,
                                const float* __restrict__ Wres_s,
                                const float* __restrict__ Wres_v,
                                float* __restrict__ res_s,
                                float* __restrict__ res_v) {
  size_t t = (size_t)blockIdx.x * 256 + threadIdx.x;
  if (t >= (size_t)kN * 256) return;
  int p = (int)(t & 3);
  size_t nf = t >> 2;
  int g = (int)(nf & 63);
  int n = (int)(nf >> 6);
  const float* W = ((p == 0) ? Wres_s : Wres_v) + ((size_t)specie[n] * 64 + g) * 64;
  const float* x = node_feats + (size_t)n * 256 + p;
  float acc = 0.f;
#pragma unroll 8
  for (int f = 0; f < 64; ++f) acc += x[f * 4] * W[f];
  acc *= 0.125f;  // inv_sf
  if (p == 0)
    res_s[(size_t)n * 64 + g] = acc;
  else
    res_v[(size_t)(p - 1) * kN * 64 + (size_t)n * 64 + g] = acc;
}

// ---------------------------------------------------------------------------
// K3: fused edge pipeline, one wave per 16 edges.
//   hidden = silu(re @ mlp_w1)           (scalar, K=8)
//   w      = hidden @ mlp_w2             (40 WMMAs, never hits HBM)
//   messages + atomic scatter (pre-scaled by 1/AVG_NEIGH)
// ---------------------------------------------------------------------------
__global__ __launch_bounds__(32) void edge_kernel(
    const float* __restrict__ vectors, const float* __restrict__ radial,
    const int* __restrict__ senders, const int* __restrict__ receivers,
    const float* __restrict__ mlp_w1, const unsigned short* __restrict__ pW2,
    const float* __restrict__ h_s, const float* __restrict__ h_v,
    float* __restrict__ agg_s, float* __restrict__ agg_v) {
  __shared__ unsigned short H[16 * 64];  // hidden, bf16
  __shared__ float wt[16 * 320];         // w tile, f32 (20KB)
  __shared__ float y1s[16][3];
  __shared__ int sn[16], rn[16];
  int lane = threadIdx.x;
  int e0 = blockIdx.x * 16;

  // first MLP layer + silu
  for (int t = lane; t < 1024; t += 32) {
    int el = t >> 6, j = t & 63;
    const float* re = radial + (size_t)(e0 + el) * 8;
    float acc = 0.f;
#pragma unroll
    for (int r = 0; r < 8; ++r) acc += re[r] * mlp_w1[r * 64 + j];
    float sig = 1.f / (1.f + __expf(-acc));
    H[t] = f2bf(acc * sig);
  }
  if (lane < 16) {
    int e = e0 + lane;
    float vx = vectors[3 * (size_t)e + 0];
    float vy = vectors[3 * (size_t)e + 1];
    float vz = vectors[3 * (size_t)e + 2];
    float inv = rsqrtf(vx * vx + vy * vy + vz * vz + 1e-12f);
    y1s[lane][0] = vx * inv;
    y1s[lane][1] = vy * inv;
    y1s[lane][2] = vz * inv;
    sn[lane] = senders[e];
    rn[lane] = receivers[e];
  }
  __syncthreads();

  // second MLP layer via WMMA: (16x64) @ (64x320)
  int lh = lane >> 4, mc = lane & 15;
  v16bf a0 = loadA(&H[mc * 64], 0, lh);
  v16bf a1 = loadA(&H[mc * 64], 1, lh);
  for (int nt = 0; nt < 20; ++nt) {
    v8f c = {};
    c = wmma_bf16(a0, loadB(pW2, 0, nt, 20, lane), c);
    c = wmma_bf16(a1, loadB(pW2, 1, nt, 20, lane), c);
    for (int r = 0; r < 8; ++r) wt[(r + 8 * lh) * 320 + nt * 16 + mc] = c[r];
  }
  __syncthreads();

  // messages + scatter
  const float cAvg = 1.0f / 16.0f;  // 1/AVG_NEIGH
  for (int t = lane; t < 1024; t += 32) {
    int el = t >> 6, f = t & 63;
    int se = sn[el], re = rn[el];
    float ss = h_s[(size_t)se * 64 + f];
    float svx = h_v[(size_t)se * 64 + f];
    float svy = h_v[(size_t)kN * 64 + (size_t)se * 64 + f];
    float svz = h_v[2 * (size_t)kN * 64 + (size_t)se * 64 + f];
    const float* w = &wt[el * 320 + f];
    float w0 = w[0], w1 = w[64], w2 = w[128], w3 = w[192], w4 = w[256];
    float yx = y1s[el][0], yy = y1s[el][1], yz = y1s[el][2];
    float dot = svx * yx + svy * yy + svz * yz;
    float ms = (w0 * ss + w1 * dot) * cAvg;
    float cx = svy * yz - svz * yy;
    float cy = svz * yx - svx * yz;
    float cz = svx * yy - svy * yx;
    float wss = w2 * ss;
    float mvx = (wss * yx + w3 * svx + w4 * cx) * cAvg;
    float mvy = (wss * yy + w3 * svy + w4 * cy) * cAvg;
    float mvz = (wss * yz + w3 * svz + w4 * cz) * cAvg;
    size_t o = (size_t)re * 64 + f;
    atomicAdd(&agg_s[o], ms);
    atomicAdd(&agg_v[o], mvx);
    atomicAdd(&agg_v[(size_t)kN * 64 + o], mvy);
    atomicAdd(&agg_v[2 * (size_t)kN * 64 + o], mvz);
  }
}

// ---------------------------------------------------------------------------
// K4: a = agg@W_out (WMMA) -> per-species polynomial (scalar) ->
//     f = p@W_lin + res (WMMA with residual preloaded into C) -> outputs.
// ---------------------------------------------------------------------------
__global__ __launch_bounds__(32) void node_out_kernel(
    const float* __restrict__ agg_s, const float* __restrict__ agg_v,
    const int* __restrict__ specie,
    const float* __restrict__ Wprod_s, const float* __restrict__ Wprod_v,
    const unsigned short* __restrict__ pWout_s, const unsigned short* __restrict__ pWout_v,
    const unsigned short* __restrict__ pWlin_s, const unsigned short* __restrict__ pWlin_v,
    const float* __restrict__ res_s, const float* __restrict__ res_v,
    const float* __restrict__ W_read, float* __restrict__ out) {
  __shared__ unsigned short A[4 * 16 * 64];  // bf16 A tiles (8KB)
  __shared__ float accf[4 * 16 * 64];        // f32 scratch (16KB)
  __shared__ int spec[16];
  int lane = threadIdx.x;
  int n0 = blockIdx.x * 16;

  for (int t = lane; t < 4096; t += 32) {
    int p = t >> 10;
    int rest = t & 1023;
    int m = rest >> 6, f = rest & 63;
    const float* src = (p == 0) ? agg_s : (agg_v + (size_t)(p - 1) * kN * 64);
    A[t] = f2bf(src[(size_t)(n0 + m) * 64 + f]);
  }
  if (lane < 16) spec[lane] = specie[n0 + lane];
  __syncthreads();

  int lh = lane >> 4, mc = lane & 15;

  // stage 1: a = agg @ W_out (inv_sf folded into pack)
  for (int p = 0; p < 4; ++p) {
    const unsigned short* pw = (p == 0) ? pWout_s : pWout_v;
    v16bf a0 = loadA(&A[p * 1024 + mc * 64], 0, lh);
    v16bf a1 = loadA(&A[p * 1024 + mc * 64], 1, lh);
    for (int nt = 0; nt < 4; ++nt) {
      v8f c = {};
      c = wmma_bf16(a0, loadB(pw, 0, nt, 4, lane), c);
      c = wmma_bf16(a1, loadB(pw, 1, nt, 4, lane), c);
      for (int r = 0; r < 8; ++r)
        accf[p * 1024 + (r + 8 * lh) * 64 + nt * 16 + mc] = c[r];
    }
  }
  __syncthreads();

  // per-species polynomial -> p tiles back into A (bf16)
  for (int t = lane; t < 1024; t += 32) {
    int m = t >> 6, f = t & 63;
    float as = accf[t];
    float ax = accf[1024 + t], ay = accf[2048 + t], az = accf[3072 + t];
    float vv = ax * ax + ay * ay + az * az;
    int sp = spec[m];
    const float* Ws = Wprod_s + (size_t)sp * 5 * 64 + f;
    const float* Wv = Wprod_v + (size_t)sp * 4 * 64 + f;
    float as2 = as * as;
    float ps = Ws[0] * as + Ws[64] * as2 + Ws[128] * as2 * as + Ws[192] * vv +
               Ws[256] * as * vv;
    float coef = Wv[0] + Wv[64] * as + Wv[128] * as2 + Wv[192] * vv;
    A[t] = f2bf(ps);
    A[1024 + t] = f2bf(coef * ax);
    A[2048 + t] = f2bf(coef * ay);
    A[3072 + t] = f2bf(coef * az);
  }
  __syncthreads();

  // stage 2: f = p @ W_lin + res ; write new_feats, stash f_s for readout
  for (int p = 0; p < 4; ++p) {
    const unsigned short* pw = (p == 0) ? pWlin_s : pWlin_v;
    const float* res = (p == 0) ? res_s : (res_v + (size_t)(p - 1) * kN * 64);
    v16bf a0 = loadA(&A[p * 1024 + mc * 64], 0, lh);
    v16bf a1 = loadA(&A[p * 1024 + mc * 64], 1, lh);
    for (int nt = 0; nt < 4; ++nt) {
      v8f c;
      for (int r = 0; r < 8; ++r)
        c[r] = res[(size_t)(n0 + r + 8 * lh) * 64 + nt * 16 + mc];
      c = wmma_bf16(a0, loadB(pw, 0, nt, 4, lane), c);
      c = wmma_bf16(a1, loadB(pw, 1, nt, 4, lane), c);
      for (int r = 0; r < 8; ++r) {
        int n = n0 + r + 8 * lh;
        int col = nt * 16 + mc;
        float val = c[r];
        out[(size_t)kN + ((size_t)n * 64 + col) * 4 + p] = val;
        if (p == 0) accf[(r + 8 * lh) * 64 + col] = val;
      }
    }
  }
  __syncthreads();

  if (lane < 16) {
    int m = lane;
    float d = 0.f;
#pragma unroll 8
    for (int f = 0; f < 64; ++f) d += accf[m * 64 + f] * W_read[f];
    out[n0 + m] = d * 0.125f;  // inv_sf
  }
}

// ---------------------------------------------------------------------------
extern "C" void kernel_launch(void* const* d_in, const int* in_sizes, int n_in,
                              void* d_out, int out_size, void* d_ws, size_t ws_size,
                              hipStream_t stream) {
  (void)in_sizes; (void)n_in; (void)out_size; (void)ws_size;
  const float* vectors    = (const float*)d_in[0];
  const float* node_feats = (const float*)d_in[1];
  const int*   specie     = (const int*)d_in[2];
  const float* radial     = (const float*)d_in[3];
  const int*   senders    = (const int*)d_in[4];
  const int*   receivers  = (const int*)d_in[5];
  const float* W_res_s    = (const float*)d_in[6];
  const float* W_res_v    = (const float*)d_in[7];
  const float* W_in_s     = (const float*)d_in[8];
  const float* W_in_v     = (const float*)d_in[9];
  const float* mlp_w1     = (const float*)d_in[10];
  const float* mlp_w2     = (const float*)d_in[11];
  const float* W_out_s    = (const float*)d_in[12];
  const float* W_out_v    = (const float*)d_in[13];
  const float* W_prod_s   = (const float*)d_in[14];
  const float* W_prod_v   = (const float*)d_in[15];
  const float* W_lin_s    = (const float*)d_in[16];
  const float* W_lin_v    = (const float*)d_in[17];
  const float* W_read     = (const float*)d_in[18];
  float* out = (float*)d_out;
  char* ws = (char*)d_ws;

  const size_t szPlane = (size_t)kN * 64 * sizeof(float);  // 5.12 MB
  size_t off = 0;
  auto take = [&](size_t bytes) {
    size_t o = off;
    off += (bytes + 255) & ~(size_t)255;
    return o;
  };
  float* h_s   = (float*)(ws + take(szPlane));
  float* h_v   = (float*)(ws + take(3 * szPlane));
  float* res_s = (float*)(ws + take(szPlane));
  float* res_v = (float*)(ws + take(3 * szPlane));
  float* agg   = (float*)(ws + take(4 * szPlane));  // agg_s then agg_v (3 planes)
  unsigned short* pWin_s  = (unsigned short*)(ws + take(8192));
  unsigned short* pWin_v  = (unsigned short*)(ws + take(8192));
  unsigned short* pWout_s = (unsigned short*)(ws + take(8192));
  unsigned short* pWout_v = (unsigned short*)(ws + take(8192));
  unsigned short* pWlin_s = (unsigned short*)(ws + take(8192));
  unsigned short* pWlin_v = (unsigned short*)(ws + take(8192));
  unsigned short* pW2     = (unsigned short*)(ws + take(40960));

  float* agg_s = agg;
  float* agg_v = agg + (size_t)kN * 64;

  const float inv_sf = 0.125f;  // 1/sqrt(64)

  // K0: pack weights (tiny)
  pack_b<<<16, 256, 0, stream>>>(W_in_s,  pWin_s,  64, inv_sf);
  pack_b<<<16, 256, 0, stream>>>(W_in_v,  pWin_v,  64, inv_sf);
  pack_b<<<16, 256, 0, stream>>>(W_out_s, pWout_s, 64, inv_sf);
  pack_b<<<16, 256, 0, stream>>>(W_out_v, pWout_v, 64, inv_sf);
  pack_b<<<16, 256, 0, stream>>>(W_lin_s, pWlin_s, 64, inv_sf);
  pack_b<<<16, 256, 0, stream>>>(W_lin_v, pWlin_v, 64, inv_sf);
  pack_b<<<80, 256, 0, stream>>>(mlp_w2,  pW2,    320, 1.0f);

  // K1: input linear transforms (WMMA)
  node_in_kernel<<<kN / 16, 32, 0, stream>>>(node_feats, pWin_s, pWin_v, h_s, h_v);

  // K2: per-species residuals (scalar)
  residual_kernel<<<(kN * 256 + 255) / 256, 256, 0, stream>>>(
      node_feats, specie, W_res_s, W_res_v, res_s, res_v);

  // zero aggregation buffers (must happen every call)
  hipMemsetAsync(agg, 0, 4 * szPlane, stream);

  // K3: fused edge MLP (WMMA) + messages + atomic scatter
  edge_kernel<<<kE / 16, 32, 0, stream>>>(vectors, radial, senders, receivers,
                                          mlp_w1, pW2, h_s, h_v, agg_s, agg_v);

  // K4: output transforms (WMMA) + polynomial + readout
  node_out_kernel<<<kN / 16, 32, 0, stream>>>(
      agg_s, agg_v, specie, W_prod_s, W_prod_v, pWout_s, pWout_v, pWlin_s,
      pWlin_v, res_s, res_v, W_read, out);
}
